// NeuralAdditiveModel_30915174596785
// MI455X (gfx1250) — matrix operand order repaired and checked by reading, not compiled, not run on hardware
//
#include <hip/hip_runtime.h>

// ---------------------------------------------------------------------------
// NAM forward for MI455X (gfx1250, wave32, WMMA).
//   h1 = relu(x[:,f,None]*W1[f] + b1[f])          (rank-1, fused into A frags)
//   h2 = relu(h1 @ W2[f] + b2[f])                 (WMMA f32_16x16x32_f16)
//   y  = h2 @ W3[f] + b3[f]                       (VALU dot + shfl reduce)
//   out[b] = bias + sum_f y[b,f]
// ---------------------------------------------------------------------------

typedef _Float16 v16h __attribute__((ext_vector_type(16)));
typedef _Float16 v8h  __attribute__((ext_vector_type(8)));
typedef float    v8f  __attribute__((ext_vector_type(8)));

namespace {
constexpr int kB        = 8192;  // batch
constexpr int kF        = 128;   // features
constexpr int kS        = 128;   // hidden 1 (K of the GEMM)
constexpr int kH        = 64;    // hidden 2 (N of the GEMM)
constexpr int kRowsWG   = 128;   // rows of B per workgroup (8 waves x 16 rows)
constexpr int kThreads  = 256;   // 8 wave32s
constexpr int kPad      = kS + 8; // 136 halves = 272B row stride: 16B aligned,
                                  // lane stride == 4 banks mod 64 -> conflict free
}

template <bool UseAtomics>
__global__ __launch_bounds__(kThreads)
void nam_forward_kernel(const float* __restrict__ x,
                        const float* __restrict__ W1,
                        const float* __restrict__ b1,
                        const float* __restrict__ W2,
                        const float* __restrict__ b2,
                        const float* __restrict__ W3,
                        const float* __restrict__ b3,
                        float* __restrict__ outOrWs)
{
    __shared__ __align__(16) _Float16 w2t[kH][kPad]; // W2[f] transposed: [h][s], f16
    __shared__ float w1s[kS];
    __shared__ float b1s[kS];
    __shared__ float xs[kRowsWG];
    __shared__ float b2s[kH];
    __shared__ float w3s[kH];
    __shared__ float b3s;

    const int f       = blockIdx.y;
    const int rowbase = blockIdx.x * kRowsWG;
    const int t       = threadIdx.x;

    // ---- cooperative stage of all per-feature parameters + x tile ----------
    if (t < kS) {
        w1s[t] = W1[f * kS + t];
        b1s[t] = b1[f * kS + t];
    }
    if (t < kRowsWG) xs[t] = x[(size_t)(rowbase + t) * kF + f];
    if (t < kH) {
        b2s[t] = b2[f * kH + t];
        w3s[t] = W3[f * kH + t];   // W3 is [F,H,1]
    }
    if (t == 0) b3s = b3[f];       // b3 is [F,1]

    // W2[f] is [s][h] row-major in global; store transposed f16 in LDS.
    const float* w2f = W2 + (size_t)f * kS * kH;
    for (int i = t; i < kS * kH; i += kThreads) {
        const int s = i >> 6;      // i / kH
        const int h = i & (kH - 1);
        w2t[h][s] = (_Float16)w2f[i];
    }
    __syncthreads();

    // ---- per-wave 16x64 = [16 x 128] @ [128 x 64] via WMMA -----------------
    const int lane = t & 31;
    const int wave = t >> 5;
    const int mIdx = lane & 15;    // row (A) / column (B,C) within tile
    const int hi   = lane >> 4;    // which half of the wave

    const float xv = xs[wave * 16 + mIdx];

    // Accumulators init = b2 broadcast (C layout: element (r,lane) has N = mIdx)
    v8f acc[4];
    #pragma unroll
    for (int nt = 0; nt < 4; ++nt) {
        const float bv = b2s[nt * 16 + mIdx];
        #pragma unroll
        for (int r = 0; r < 8; ++r) acc[nt][r] = bv;
    }

    #pragma unroll
    for (int kt = 0; kt < 4; ++kt) {
        const int k0 = kt * 32;

        // A fragment (16-bit A 16x32 layout, ISA 7.12.2):
        //   lanes 0-15 : VGPR0-3 -> K 0..7,   VGPR4-7 -> K 16..23
        //   lanes 16-31: VGPR0-3 -> K 8..15,  VGPR4-7 -> K 24..31
        // Fused layer 1: a = f16(relu(x*W1[k] + b1[k]))
        v16h a;
        #pragma unroll
        for (int j = 0; j < 16; ++j) {
            const int v    = j >> 1;
            const int p    = j & 1;
            const int koff = ((v & 4) ? 16 : 0) + (hi << 3) + ((v & 3) << 1) + p;
            const int k    = k0 + koff;
            float h1v = fmaf(xv, w1s[k], b1s[k]);
            h1v = fmaxf(h1v, 0.0f);
            a[j] = (_Float16)h1v;
        }

        #pragma unroll
        for (int nt = 0; nt < 4; ++nt) {
            // B fragment (16-bit B 32x16): lane = N; lanes 0-15 hold K 0..7, 16..23,
            // lanes 16-31 hold K 8..15, 24..31 (same K order as A).
            // -> two 8-half runs from the transposed LDS image.
            const int N = nt * 16 + mIdx;
            const _Float16* bp = &w2t[N][k0 + (hi << 3)];
            const v8h blo = *(const v8h*)(bp);       // 16B aligned
            const v8h bhi = *(const v8h*)(bp + 16);
            v16h bb;
            #pragma unroll
            for (int j = 0; j < 8; ++j) { bb[j] = blo[j]; bb[j + 8] = bhi[j]; }

            acc[nt] = __builtin_amdgcn_wmma_f32_16x16x32_f16(
                /*neg_a=*/false, a, /*neg_b=*/false, bb,
                /*c_mod=*/(short)0, acc[nt],
                /*reuse_a=*/false, /*reuse_b=*/false);
        }
    }

    // ---- layer 3: relu(h2) . W3  (C layout: reg r holds row r + 8*hi) ------
    float part[8];
    #pragma unroll
    for (int r = 0; r < 8; ++r) part[r] = 0.0f;
    #pragma unroll
    for (int nt = 0; nt < 4; ++nt) {
        const float w3v = w3s[nt * 16 + mIdx];
        #pragma unroll
        for (int r = 0; r < 8; ++r)
            part[r] = fmaf(fmaxf(acc[nt][r], 0.0f), w3v, part[r]);
    }
    // Reduce the 64 h-terms across the 16 lanes of each half-wave.
    #pragma unroll
    for (int off = 1; off < 16; off <<= 1) {
        #pragma unroll
        for (int r = 0; r < 8; ++r)
            part[r] += __shfl_xor(part[r], off, 32);
    }

    if (mIdx == 0) {
        const int row0 = rowbase + wave * 16 + hi * 8;
        #pragma unroll
        for (int r = 0; r < 8; ++r) {
            const float y = part[r] + b3s;
            if constexpr (UseAtomics) {
                atomicAdd(&outOrWs[row0 + r], y);
            } else {
                outOrWs[(size_t)f * kB + row0 + r] = y;
            }
        }
    }
}

__global__ __launch_bounds__(256)
void nam_reduce_kernel(const float* __restrict__ ws,
                       const float* __restrict__ bias,
                       float* __restrict__ out)
{
    const int b = blockIdx.x * blockDim.x + threadIdx.x;
    float s = bias[0];
    #pragma unroll 8
    for (int f = 0; f < kF; ++f) s += ws[(size_t)f * kB + b]; // coalesced per f
    out[b] = s;
}

__global__ __launch_bounds__(256)
void nam_init_out(float* __restrict__ out, const float* __restrict__ bias)
{
    const int b = blockIdx.x * blockDim.x + threadIdx.x;
    out[b] = bias[0];
}

extern "C" void kernel_launch(void* const* d_in, const int* in_sizes, int n_in,
                              void* d_out, int out_size, void* d_ws, size_t ws_size,
                              hipStream_t stream)
{
    (void)in_sizes; (void)n_in; (void)out_size;
    const float* x    = (const float*)d_in[0];
    const float* W1   = (const float*)d_in[1];
    const float* b1   = (const float*)d_in[2];
    const float* W2   = (const float*)d_in[3];
    const float* b2   = (const float*)d_in[4];
    const float* W3   = (const float*)d_in[5];
    const float* b3   = (const float*)d_in[6];
    const float* bias = (const float*)d_in[7];
    float* out = (float*)d_out;

    const dim3 grid(kB / kRowsWG, kF);   // (64, 128)
    const size_t wsNeed = (size_t)kF * kB * sizeof(float);  // 4 MB

    if (ws_size >= wsNeed) {
        // Deterministic path: per-feature partial rows -> workspace -> reduce.
        float* ws = (float*)d_ws;
        nam_forward_kernel<false><<<grid, kThreads, 0, stream>>>(
            x, W1, b1, W2, b2, W3, b3, ws);
        nam_reduce_kernel<<<kB / 256, 256, 0, stream>>>(ws, bias, out);
    } else {
        // Fallback: init out with bias, accumulate features via f32 atomics.
        nam_init_out<<<kB / 256, 256, 0, stream>>>(out, bias);
        nam_forward_kernel<true><<<grid, kThreads, 0, stream>>>(
            x, W1, b1, W2, b2, W3, b3, out);
    }
}
